// SharedNet_40114994545046
// MI455X (gfx1250) — matrix operand order repaired and checked by reading, not compiled, not run on hardware
//
#include <hip/hip_runtime.h>
#include <hip/hip_bf16.h>
#include <stdint.h>

typedef _Float16 h16;
typedef __attribute__((ext_vector_type(16))) _Float16 v16h;
typedef __attribute__((ext_vector_type(8)))  _Float16 v8h;
typedef __attribute__((ext_vector_type(8)))  float    v8f;
typedef __attribute__((ext_vector_type(4)))  float    v4f;

union V16u { v16h v; v8h h[2]; };
union V8hu { v8h v; h16 e[8]; };

// ---------------------------------------------------------------------------
// Core WMMA implicit-im2col conv/GEMM. Branch-free, division-free hot loop.
//   A   : fp16 weights [M][KH*KW*C] row-major, K-order (kh,kw,ci)
//   src : fp16 NHWC, HALO-PADDED to [(W+2P) x (W+2P)] per sample: interior
//         pixel (y,x) lives at padded (y+P, x+P); halo pre-zeroed. For
//         KH=KW=1 (P=0) this is a plain unpadded [pix][C] GEMM operand.
//   out : fp16 NHWC [pix][M] (unpadded)
// W, C, KH, KW are compile-time: W is a power of two so pixel decomposition
// is shifts/masks, and all tap/chunk offsets fold into load immediates.
// Each wave computes NB 16x16 D tiles sharing one A fragment (A reuse).
// Lane layouts used here (v_wmma_f32_16x16x32_f16; contiguous k, same for A and B):
//   A 16x32 : halves e0..15 -> k=kt+hi*16+e
//   B 32x16 : halves e0..15 -> k=kt+hi*16+e
//   D 16x16 : VGPR r -> (m = mBase+hi*8+r, pixel = tileBase+(lane&15))
// ---------------------------------------------------------------------------
template<int KH, int KW, int C, int W, int NB>
__global__ __launch_bounds__(256)
void k_wmma_conv(const h16* __restrict__ A, const h16* __restrict__ src,
                 const float* __restrict__ bias, h16* __restrict__ out,
                 int M, int Npix, int sampleStrideA, int sampleStrideBias, int relu)
{
    constexpr int PAD = (KH - 1) / 2;
    constexpr int WP  = W + 2 * PAD;       // padded source width
    constexpr int HW  = W * W;             // output pixels per sample (pow2)
    constexpr int K   = KH * KW * C;

    const int lane = threadIdx.x;          // 0..31 (wave32)
    const int hi   = lane >> 4;
    const int ln   = lane & 15;
    const int grp  = blockIdx.x * blockDim.y + threadIdx.y;
    const int mBase = blockIdx.y * 16;
    if (grp * (16 * NB) >= Npix) return;   // wave-uniform guard (EXEC full)

    const int pix0   = grp * (16 * NB) + ln;
    const int sample = pix0 / HW;          // pow2 -> shift

    const h16* tp[NB];
#pragma unroll
    for (int t = 0; t < NB; ++t) {
        const int rem = (pix0 + t * 16) & (HW - 1);
        const int py  = rem / W;           // pow2 -> shift
        const int px  = rem & (W - 1);
        tp[t] = src + (long)sample * (WP * WP * C) + ((long)py * WP + px) * C + hi * 16;
    }
    const h16* Arow = A + (long)sample * sampleStrideA + (long)(mBase + ln) * K;

    v8f acc[NB];
#pragma unroll
    for (int t = 0; t < NB; ++t) acc[t] = (v8f){};

    for (int kh = 0; kh < KH; ++kh) {      // small trip; compiler may unroll
        const int sRow = kh * (WP * C);
        const int aRow = kh * (KW * C);
#pragma unroll
        for (int kw = 0; kw < KW; ++kw) {
            const int sOff = sRow + kw * C;
            const int aOff = aRow + kw * C;
#pragma unroll
            for (int ci = 0; ci < C; ci += 32) {
                V16u av;
                av.h[0] = *(const v8h*)(Arow + aOff + ci + hi * 16);
                av.h[1] = *(const v8h*)(Arow + aOff + ci + hi * 16 + 8);
#pragma unroll
                for (int t = 0; t < NB; ++t) {
                    V16u bv;
                    bv.h[0] = *(const v8h*)(tp[t] + sOff + ci);
                    bv.h[1] = *(const v8h*)(tp[t] + sOff + ci + 8);
                    acc[t] = __builtin_amdgcn_wmma_f32_16x16x32_f16(
                                 false, av.v, false, bv.v, (short)0, acc[t],
                                 false, false);
                }
            }
        }
    }

    const float* bp = bias + (long)sample * sampleStrideBias + mBase + hi * 8;
    const v4f b0 = *(const v4f*)bp;
    const v4f b1 = *(const v4f*)(bp + 4);
#pragma unroll
    for (int t = 0; t < NB; ++t) {
        V8hu ov;
#pragma unroll
        for (int r = 0; r < 8; ++r) {
            float v = acc[t][r] + (r < 4 ? b0[r & 3] : b1[r & 3]);
            if (relu) v = fmaxf(v, 0.f);
            ov.e[r] = (h16)v;
        }
        const int pix = pix0 + t * 16;
        *(v8h*)(out + (long)pix * M + mBase + hi * 8) = ov.v;
    }
}

// ---------------------------------------------------------------------------
// Support kernels
// ---------------------------------------------------------------------------

__global__ void k_zero(uint32_t* __restrict__ p, int total)
{
    int i = blockIdx.x * blockDim.x + threadIdx.x;
    if (i < total) p[i] = 0u;
}

// conv1 im2col: x fp32 NCHW (16,3,128,128) -> B1 fp16 [262144][96], k=(ci*25+kh*5+kw), pad 75->96
__global__ void k_im2col_conv1(const float* __restrict__ x, h16* __restrict__ B1, int total)
{
    int i = blockIdx.x * blockDim.x + threadIdx.x;
    if (i >= total) return;
    int pix = i / 96, k = i - pix * 96;
    float v = 0.f;
    if (k < 75) {
        int n = pix >> 14, r = pix & 16383, py = r >> 7, px = r & 127;
        int ci = k / 25, rr = k - ci * 25, kh = rr / 5, kw = rr - kh * 5;
        int y = py + kh - 2, xx = px + kw - 2;
        if ((unsigned)y < 128u && (unsigned)xx < 128u)
            v = x[(((long)n * 3 + ci) * 128 + y) * 128 + xx];
    }
    B1[i] = (h16)v;
}

// fp32 [Mr][Kr] -> fp16 [Mpad][Kpad] zero-padded (also covers w1 as [192][75]->[192][96])
__global__ void k_pack_dense(const float* __restrict__ w, h16* __restrict__ A,
                             int Kpad, int Mr, int Kr, int total)
{
    int i = blockIdx.x * blockDim.x + threadIdx.x;
    if (i >= total) return;
    int co = i / Kpad, k = i - co * Kpad;
    float v = (co < Mr && k < Kr) ? w[(long)co * Kr + k] : 0.f;
    A[i] = (h16)v;
}

// aw1 (96,96,3,3) -> fp16 [96][864] with k = (kh*3+kw)*96 + ci
__global__ void k_pack_aw1(const float* __restrict__ w, h16* __restrict__ A, int total)
{
    int i = blockIdx.x * blockDim.x + threadIdx.x;
    if (i >= total) return;
    int co = i / 864, k = i - co * 864;
    int tap = k / 96, ci = k - tap * 96, kh = tap / 3, kw = tap - kh * 3;
    A[i] = (h16)w[(((long)co * 96 + ci) * 3 + kh) * 3 + kw];
}

__global__ void k_pad_bias16(const float* __restrict__ b, float* __restrict__ o)
{
    int t = threadIdx.x;
    if (t < 16) o[t] = (t < 10) ? b[t] : 0.f;
}

// hypernet weights: Ah[j][s][co][(kh*5+kw)*96+ci] = hc(j,s)*lin_w[flat] + lin_b[flat]
__global__ void k_pack_hyper_w(const float* __restrict__ lin_w, const float* __restrict__ lin_b,
                               const float* __restrict__ sig, h16* __restrict__ Ah, int total)
{
    int i = blockIdx.x * blockDim.x + threadIdx.x;
    if (i >= total) return;
    int k = i % 2400; int t = i / 2400;
    int co = t % 96;  t /= 96;
    int s = t & 15, j = t >> 4;
    float sb = sig[s];
    float h  = j ? sb : 1.f - sb;
    float hc = (h > 1e-4f ? 0.5f : 0.f) + h * (1.f / 64.f);
    int tap = k / 96, ci = k - tap * 96, kh = tap / 5, kw = tap - kh * 5;
    long flat = (long)co * 2400 + (ci * 25 + kh * 5 + kw);
    Ah[i] = (h16)(hc * lin_w[flat] + lin_b[flat]);
}

__global__ void k_pack_hyper_b(const float* __restrict__ lin_w, const float* __restrict__ lin_b,
                               const float* __restrict__ sig, float* __restrict__ bh, int total)
{
    int i = blockIdx.x * blockDim.x + threadIdx.x;
    if (i >= total) return;
    int co = i % 96; int t = i / 96;
    int s = t & 15, j = t >> 4;
    float sb = sig[s];
    float h  = j ? sb : 1.f - sb;
    float hc = (h > 1e-4f ? 0.5f : 0.f) + h * (1.f / 64.f);
    bh[i] = hc * lin_w[230400 + co] + lin_b[230400 + co];
}

// maxpool 3x3 s2 p1 on NHWC fp16 (128x128 -> 64x64, C=96), writes into
// halo-padded [16][68][68][96] at (+2,+2); halo pre-zeroed by k_zero.
__global__ void k_maxpool_pad(const h16* __restrict__ in, h16* __restrict__ out, int total)
{
    int i = blockIdx.x * blockDim.x + threadIdx.x;
    if (i >= total) return;
    int c = i % 96; int t = i / 96;
    int n = t >> 12, r = t & 4095, oy = r >> 6, ox = r & 63;
    float m = -1e30f;
#pragma unroll
    for (int dy = 0; dy < 3; ++dy)
#pragma unroll
        for (int dx = 0; dx < 3; ++dx) {
            int y = 2 * oy - 1 + dy, x = 2 * ox - 1 + dx;
            if ((unsigned)y < 128u && (unsigned)x < 128u)
                m = fmaxf(m, (float)in[(((long)n * 128 + y) * 128 + x) * 96 + c]);
        }
    out[(((long)n * 68 + oy + 2) * 68 + (ox + 2)) * 96 + c] = (h16)m;
}

// avgpool 3x3 s2 p1 (64x64 -> 32x32, C=96), unpadded in/out, sum/9
__global__ void k_avgpool(const h16* __restrict__ in, h16* __restrict__ out, int total)
{
    int i = blockIdx.x * blockDim.x + threadIdx.x;
    if (i >= total) return;
    int c = i % 96; int t = i / 96;
    int n = t >> 10, r = t & 1023, oy = r >> 5, ox = r & 31;
    float s = 0.f;
#pragma unroll
    for (int dy = 0; dy < 3; ++dy)
#pragma unroll
        for (int dx = 0; dx < 3; ++dx) {
            int y = 2 * oy - 1 + dy, x = 2 * ox - 1 + dx;
            if ((unsigned)y < 64u && (unsigned)x < 64u)
                s += (float)in[(((long)n * 64 + y) * 64 + x) * 96 + c];
        }
    out[i] = (h16)(s * (1.f / 9.f));
}

// spatial mean of padded p [16][68][68][96] interior -> pooled [16][96] fp32
__global__ void k_mean(const h16* __restrict__ p, float* __restrict__ pooled, int total)
{
    int i = blockIdx.x * blockDim.x + threadIdx.x;
    if (i >= total) return;
    int n = i / 96, c = i - n * 96;
    float s = 0.f;
    for (int y = 0; y < 64; ++y)
        for (int x = 0; x < 64; ++x)
            s += (float)p[(((long)n * 68 + y + 2) * 68 + (x + 2)) * 96 + c];
    pooled[i] = s * (1.f / 4096.f);
}

// gating: logits = pooled @ sel_w.T + sel_b ; softmax ; sig[0..16)=sb, sig[16..48)=sr
__global__ void k_sel(const float* __restrict__ pooled, const float* __restrict__ sel_w,
                      const float* __restrict__ sel_b, float* __restrict__ sig)
{
    int n = threadIdx.x;
    if (n >= 16) return;
    float l0 = sel_b[0], l1 = sel_b[1];
    for (int c = 0; c < 96; ++c) {
        float pv = pooled[n * 96 + c];
        l0 += pv * sel_w[c];
        l1 += pv * sel_w[96 + c];
    }
    float mx = fmaxf(l0, l1);
    float e0 = __expf(l0 - mx), e1 = __expf(l1 - mx);
    float inv = 1.f / (e0 + e1);
    sig[n] = e1 * inv;
    sig[16 + n * 2 + 0] = e0 * inv;
    sig[16 + n * 2 + 1] = e1 * inv;
}

// xm_pad[(n,oy+1,ox+1,c)] = (1-sb)*x0 + sb*x1 ; x0/x1 unpadded 32x32x96
__global__ void k_mix_pad(const h16* __restrict__ x0, const h16* __restrict__ x1,
                          const float* __restrict__ sig, h16* __restrict__ xm, int total)
{
    int i = blockIdx.x * blockDim.x + threadIdx.x;
    if (i >= total) return;
    int c = i % 96; int t = i / 96;
    int n = t >> 10, r = t & 1023, oy = r >> 5, ox = r & 31;
    float s = sig[n];
    float v = (1.f - s) * (float)x0[i] + s * (float)x1[i];
    xm[(((long)n * 34 + oy + 1) * 34 + (ox + 1)) * 96 + c] = (h16)v;
}

// y3 NHWC fp16 [16*1024][16] -> d_out fp32 NCHW (16,10,32,32), then sigmas
__global__ void k_finalize(const h16* __restrict__ y3, const float* __restrict__ sig,
                           float* __restrict__ dst, int total)
{
    int i = blockIdx.x * blockDim.x + threadIdx.x;
    if (i >= total) return;
    if (i < 163840) {
        int n = i / 10240, r = i - n * 10240, c = r >> 10, q = r & 1023;
        dst[i] = (float)y3[((long)n * 1024 + q) * 16 + c];
    } else {
        int j = i - 163840;
        if (j < 16)       dst[i] = sig[j];             // sigmas_b
        else if (j < 48)  dst[i] = sig[16 + (j - 16)]; // sigmas_r (16,2)
        else              dst[i] = sig[j - 48];        // sigmas_b_r
    }
}

// ---------------------------------------------------------------------------
extern "C" void kernel_launch(void* const* d_in, const int* in_sizes, int n_in,
                              void* d_out, int out_size, void* d_ws, size_t ws_size,
                              hipStream_t stream)
{
    const float* x     = (const float*)d_in[0];
    const float* w1    = (const float*)d_in[1];
    const float* b1    = (const float*)d_in[2];
    const float* w2    = (const float*)d_in[3];
    const float* b2    = (const float*)d_in[4];
    const float* w3    = (const float*)d_in[5];
    const float* b3    = (const float*)d_in[6];
    const float* sel_w = (const float*)d_in[7];
    const float* sel_b = (const float*)d_in[8];
    const float* lin_w = (const float*)d_in[9];
    const float* lin_b = (const float*)d_in[10];
    const float* hc1w  = (const float*)d_in[11];
    const float* hc1b  = (const float*)d_in[12];
    const float* hc2w  = (const float*)d_in[13];
    const float* hc2b  = (const float*)d_in[14];
    const float* aw1   = (const float*)d_in[15];
    const float* ab1   = (const float*)d_in[16];
    const float* aw2   = (const float*)d_in[17];
    const float* ab2   = (const float*)d_in[18];
    const float* aw3   = (const float*)d_in[19];
    const float* ab3   = (const float*)d_in[20];
    // d_in[21] = binarize (unused by reference math)

    // ---- workspace layout (region-overlaid; ~286 MB peak) ----
    char* ws = (char*)d_ws;
    size_t off = 0;
    auto take = [&](size_t bytes) -> char* {
        char* ptr = ws + off;
        off += (bytes + 255) & ~(size_t)255;
        return ptr;
    };
    char* regA = take(50331648);   // B1 im2col; later hypernet/tail buffers
    char* regB = take(100663296);  // act1; later hyper weights
    char* regC = take(83886080);   // act2; later p_pad/pooled/sig
    char* regD = take(50331648);   // act3
    h16* w1p  = (h16*)take(192 * 96 * 2);
    h16* w2p  = (h16*)take(160 * 192 * 2);
    h16* w3p  = (h16*)take(96 * 160 * 2);
    h16* hc1p = (h16*)take(96 * 96 * 2);
    h16* hc2p = (h16*)take(96 * 96 * 2);
    h16* aw1p = (h16*)take(96 * 864 * 2);
    h16* aw2p = (h16*)take(96 * 96 * 2);
    h16* aw3p = (h16*)take(16 * 96 * 2);

    h16* B1   = (h16*)regA;
    h16* act1 = (h16*)regB;
    h16* act2 = (h16*)regC;
    h16* act3 = (h16*)regD;
    // regA reuse after backbone:
    h16* t0     = (h16*)(regA + 0);          // [65536][96]
    h16* t1     = (h16*)(regA + 12582912);   // [65536][96]
    h16* x0     = (h16*)(regA + 25165824);   // [16384][96]
    h16* x1     = (h16*)(regA + 28311552);   // [16384][96]
    h16* xm_pad = (h16*)(regA + 31457280);   // [16][34][34][96]
    h16* y1     = (h16*)(regA + 35008512);   // [16384][96]
    h16* y2     = (h16*)(regA + 38154240);   // [16384][96]
    h16* y3     = (h16*)(regA + 41299968);   // [16384][16]
    // regB reuse:
    h16*   Ah    = (h16*)regB;                 // [2][16][96][2400] fp16
    float* biash = (float*)(regB + 14745600);  // [2][16][96]
    float* b3pad = (float*)(regB + 14758144);  // [16]
    // regC reuse:
    h16*   p_pad  = (h16*)regC;                // [16][68][68][96] fp16 (halo 2)
    float* pooled = (float*)(regC + 14204928); // [16][96]
    float* sig    = (float*)(regC + 14211072); // [48]: sb + sr

    auto cdiv = [](int a, int b) { return (a + b - 1) / b; };
    const dim3 wblk(32, 8);   // 8 wave32 waves per block; each wave: 4 tiles

    // ---- 1. pack weights (fp32 -> fp16, zero-padded) ----
    k_pack_dense<<<cdiv(192 * 96, 256), 256, 0, stream>>>(w1, w1p, 96, 192, 75, 192 * 96);
    k_pack_dense<<<cdiv(160 * 192, 256), 256, 0, stream>>>(w2, w2p, 192, 160, 192, 160 * 192);
    k_pack_dense<<<cdiv(96 * 160, 256), 256, 0, stream>>>(w3, w3p, 160, 96, 160, 96 * 160);
    k_pack_dense<<<cdiv(96 * 96, 256), 256, 0, stream>>>(hc1w, hc1p, 96, 96, 96, 96 * 96);
    k_pack_dense<<<cdiv(96 * 96, 256), 256, 0, stream>>>(hc2w, hc2p, 96, 96, 96, 96 * 96);
    k_pack_dense<<<cdiv(96 * 96, 256), 256, 0, stream>>>(aw2, aw2p, 96, 96, 96, 96 * 96);
    k_pack_dense<<<cdiv(16 * 96, 256), 256, 0, stream>>>(aw3, aw3p, 96, 10, 96, 16 * 96);
    k_pack_aw1<<<cdiv(96 * 864, 256), 256, 0, stream>>>(aw1, aw1p, 96 * 864);
    k_pad_bias16<<<1, 16, 0, stream>>>(ab3, b3pad);

    // ---- 2. backbone (262144 pixels -> 4096 groups of 4 tiles) ----
    k_im2col_conv1<<<cdiv(262144 * 96, 256), 256, 0, stream>>>(x, B1, 262144 * 96);
    k_wmma_conv<1, 1, 96, 128, 4><<<dim3(512, 12), wblk, 0, stream>>>(
        w1p, B1, b1, act1, 192, 262144, 0, 0, 1);
    k_wmma_conv<1, 1, 192, 128, 4><<<dim3(512, 10), wblk, 0, stream>>>(
        w2p, act1, b2, act2, 160, 262144, 0, 0, 1);
    k_wmma_conv<1, 1, 160, 128, 4><<<dim3(512, 6), wblk, 0, stream>>>(
        w3p, act2, b3, act3, 96, 262144, 0, 0, 1);

    // ---- 3. pooling + gating (p_pad: zero halo then write interior) ----
    k_zero<<<cdiv(3551232, 256), 256, 0, stream>>>((uint32_t*)p_pad, 3551232);
    k_maxpool_pad<<<cdiv(16 * 4096 * 96, 256), 256, 0, stream>>>(act3, p_pad, 16 * 4096 * 96);
    k_mean<<<cdiv(16 * 96, 256), 256, 0, stream>>>(p_pad, pooled, 16 * 96);
    k_sel<<<1, 32, 0, stream>>>(pooled, sel_w, sel_b, sig);

    // ---- 4. hypernet weight generation ----
    k_pack_hyper_w<<<cdiv(7372800, 256), 256, 0, stream>>>(lin_w, lin_b, sig, Ah, 7372800);
    k_pack_hyper_b<<<cdiv(3072, 256), 256, 0, stream>>>(lin_w, lin_b, sig, biash, 3072);

    // ---- 5. two hyper branches (65536 pixels -> 1024 groups) ----
    for (int j = 0; j < 2; ++j) {
        const h16*   Aj = Ah + (size_t)j * 16 * 96 * 2400;
        const float* bj = biash + (size_t)j * 16 * 96;
        h16*         xo = j ? x1 : x0;
        k_wmma_conv<5, 5, 96, 64, 4><<<dim3(128, 6), wblk, 0, stream>>>(
            Aj, p_pad, bj, t0, 96, 65536, 96 * 2400, 96, 1);
        k_wmma_conv<1, 1, 96, 64, 4><<<dim3(128, 6), wblk, 0, stream>>>(
            hc1p, t0, hc1b, t1, 96, 65536, 0, 0, 1);
        k_wmma_conv<1, 1, 96, 64, 4><<<dim3(128, 6), wblk, 0, stream>>>(
            hc2p, t1, hc2b, t0, 96, 65536, 0, 0, 1);
        k_avgpool<<<cdiv(16 * 1024 * 96, 256), 256, 0, stream>>>(t0, xo, 16 * 1024 * 96);
    }

    // ---- 6. mix + tail convs (16384 pixels -> 256 groups) ----
    k_zero<<<cdiv(887808, 256), 256, 0, stream>>>((uint32_t*)xm_pad, 887808);
    k_mix_pad<<<cdiv(16384 * 96, 256), 256, 0, stream>>>(x0, x1, sig, xm_pad, 16384 * 96);
    k_wmma_conv<3, 3, 96, 32, 4><<<dim3(32, 6), wblk, 0, stream>>>(
        aw1p, xm_pad, ab1, y1, 96, 16384, 0, 0, 1);
    k_wmma_conv<1, 1, 96, 32, 4><<<dim3(32, 6), wblk, 0, stream>>>(
        aw2p, y1, ab2, y2, 96, 16384, 0, 0, 1);
    k_wmma_conv<1, 1, 96, 32, 4><<<dim3(32, 1), wblk, 0, stream>>>(
        aw3p, y2, b3pad, y3, 16, 16384, 0, 0, 0);

    // ---- 7. finalize outputs (x NCHW fp32 + sigmas) ----
    k_finalize<<<cdiv(163904, 256), 256, 0, stream>>>(y3, sig, (float*)d_out, 163904);
    (void)in_sizes; (void)n_in; (void)out_size; (void)ws_size;
}
